// SchemaAttentionLayer_32744830664927
// MI455X (gfx1250) — compile-verified, matrix operand order checked by reading, not run on hardware
//
#include <hip/hip_runtime.h>

#define N_ROWS 200000
#define IN_F   256
#define OUT_F  128
#define ALPHA  0.2f
#define KPAD   264   // padded K stride (halves): 256 + 8 -> 528B row stride

typedef __attribute__((ext_vector_type(16))) _Float16 v16h;
typedef __attribute__((ext_vector_type(8)))  _Float16 v8h;
typedef __attribute__((ext_vector_type(8)))  float    v8f;
typedef __attribute__((ext_vector_type(4)))  float    v4f;

// ---------------------------------------------------------------------------
// Prep: u[k] = sum_c W[k,c] * s[c] ; v[k] = sum_c W[k,c] * s[128+c]
// ---------------------------------------------------------------------------
__global__ void gat_prep(const float* __restrict__ W, const float* __restrict__ s,
                         float* __restrict__ uv) {
  const int k = threadIdx.x;  // 0..255
  const float* wrow = W + k * OUT_F;
  float su = 0.f, sv = 0.f;
  for (int c = 0; c < OUT_F; ++c) {
    const float w = wrow[c];
    su += w * s[c];
    sv += w * s[OUT_F + c];
  }
  uv[k]       = su;
  uv[256 + k] = sv;
}

// ---------------------------------------------------------------------------
// Main: per 128-row tile -- scores + softmax in fp32, blended row -> f16,
// single projection GEMM via V_WMMA_F32_16X16X32_F16.
// ---------------------------------------------------------------------------
__global__ void __launch_bounds__(256) gat_main(
    const float* __restrict__ t_in, const float* __restrict__ o1m,
    const float* __restrict__ o2m, const float* __restrict__ W,
    const float* __restrict__ uv, float* __restrict__ out) {
  extern __shared__ char smem_raw[];
  _Float16* s_mat = (_Float16*)smem_raw;                       // 128*KPAD halves (Wt, then Y)
  float*    s_uv  = (float*)(smem_raw + 128 * KPAD * 2);       // 512 floats

  const int tid  = threadIdx.x;
  const int lane = tid & 31;
  const int wv   = tid >> 5;        // wave id == output-column tile jt
  const int lrow = lane & 15;
  const int hh   = lane >> 4;       // K-half selector (A/B 16-bit layout)

  // u,v -> LDS
  s_uv[tid]       = uv[tid];
  s_uv[256 + tid] = uv[256 + tid];

  // W (f32, [256][128] row-major) -> LDS as f16, K-transposed: Wt[c][k]
  for (int i = 0; i < 128; ++i) {
    const int idx = tid + 256 * i;   // over 256*128 elements
    const int k = idx >> 7;
    const int c = idx & 127;
    s_mat[c * KPAD + k] = (_Float16)W[idx];
  }
  __syncthreads();

  // Extract this wave's B fragments (cols [16*wv, 16*wv+16), all 8 K-blocks)
  v16h bfrag[8];
  {
    const int col = 16 * wv + lrow;
    const _Float16* base = s_mat + col * KPAD + hh * 8;
#pragma unroll
    for (int kb = 0; kb < 8; ++kb) {
      v8h lo = *(const v8h*)(base + 32 * kb);        // K = 32kb+8h   .. +7
      v8h hi = *(const v8h*)(base + 32 * kb + 16);   // K = 32kb+8h+16.. +23
      bfrag[kb] = __builtin_shufflevector(lo, hi, 0, 1, 2, 3, 4, 5, 6, 7,
                                          8, 9, 10, 11, 12, 13, 14, 15);
    }
  }
  __syncthreads();   // Wt dead; s_mat reused as Y

  const long tileBase  = (long)blockIdx.x * 128;
  const bool fullTile  = (tileBase + 128) <= (long)N_ROWS;  // block-uniform

  // ---- Producer: scores, softmax, blended f16 row -> LDS -------------------
  {
    const int  r   = 16 * wv + lrow;          // row within tile for this lane
    const long g   = tileBase + r;
    const bool inb = fullTile || (g < N_ROWS);
    const float* tr = t_in + g * IN_F;
    const float* r1 = o1m + g * IN_F;
    const float* r2 = o2m + g * IN_F;

    float du = 0.f, dvt = 0.f, dv1 = 0.f, dv2 = 0.f;
    if (inb) {
#pragma unroll 4
      for (int i = 0; i < 16; ++i) {
        const int K0 = 16 * i + 8 * hh;       // this lane's K-half coverage
        const v4f t0 = *(const v4f*)(tr + K0);
        const v4f t1 = *(const v4f*)(tr + K0 + 4);
        const v4f p0 = *(const v4f*)(r1 + K0);
        const v4f p1 = *(const v4f*)(r1 + K0 + 4);
        const v4f q0 = *(const v4f*)(r2 + K0);
        const v4f q1 = *(const v4f*)(r2 + K0 + 4);
        const v4f u0 = *(const v4f*)(s_uv + K0);
        const v4f u1 = *(const v4f*)(s_uv + K0 + 4);
        const v4f v0 = *(const v4f*)(s_uv + 256 + K0);
        const v4f v1 = *(const v4f*)(s_uv + 256 + K0 + 4);
#pragma unroll
        for (int j = 0; j < 4; ++j) {
          du  += t0[j] * u0[j] + t1[j] * u1[j];
          dvt += t0[j] * v0[j] + t1[j] * v1[j];
          dv1 += p0[j] * v0[j] + p1[j] * v1[j];
          dv2 += q0[j] * v0[j] + q1[j] * v1[j];
        }
      }
    }
    // combine the two K-halves of each row
    du  += __shfl_xor(du, 16);
    dvt += __shfl_xor(dvt, 16);
    dv1 += __shfl_xor(dv1, 16);
    dv2 += __shfl_xor(dv2, 16);

    float e0 = du + dvt, e1 = du + dv1, e2 = du + dv2;
    e0 = e0 > 0.f ? e0 : ALPHA * e0;
    e1 = e1 > 0.f ? e1 : ALPHA * e1;
    e2 = e2 > 0.f ? e2 : ALPHA * e2;
    const float mx = fmaxf(e0, fmaxf(e1, e2));
    const float w0 = __expf(e0 - mx), w1 = __expf(e1 - mx), w2 = __expf(e2 - mx);
    const float inv = 1.f / (w0 + w1 + w2);
    const float a0 = w0 * inv, a1 = w1 * inv, a2 = w2 * inv;

    _Float16* yrow = s_mat + r * KPAD;
    if (inb) {
#pragma unroll 4
      for (int i = 0; i < 16; ++i) {
        const int K0 = 16 * i + 8 * hh;
        const v4f t0 = *(const v4f*)(tr + K0);
        const v4f t1 = *(const v4f*)(tr + K0 + 4);
        const v4f p0 = *(const v4f*)(r1 + K0);
        const v4f p1 = *(const v4f*)(r1 + K0 + 4);
        const v4f q0 = *(const v4f*)(r2 + K0);
        const v4f q1 = *(const v4f*)(r2 + K0 + 4);
        v8h yh;
#pragma unroll
        for (int j = 0; j < 4; ++j) {
          yh[j]     = (_Float16)(a0 * t0[j] + a1 * p0[j] + a2 * q0[j]);
          yh[4 + j] = (_Float16)(a0 * t1[j] + a1 * p1[j] + a2 * q1[j]);
        }
        *(v8h*)(yrow + K0) = yh;
      }
    } else {
      const v8h z = {(_Float16)0, (_Float16)0, (_Float16)0, (_Float16)0,
                     (_Float16)0, (_Float16)0, (_Float16)0, (_Float16)0};
#pragma unroll
      for (int i = 0; i < 16; ++i) *(v8h*)(yrow + 16 * i + 8 * hh) = z;
    }
  }
  __syncthreads();

  // ---- Consumer: wave wv computes out[:, 16wv:16wv+16] for all 8 sub-tiles -
  const int colg = 16 * wv + lrow;
  for (int rt = 0; rt < 8; ++rt) {
    v8f acc = {};
    const _Float16* arow = s_mat + (rt * 16 + lrow) * KPAD + hh * 8;
#pragma unroll
    for (int kb = 0; kb < 8; ++kb) {
      v8h lo = *(const v8h*)(arow + 32 * kb);
      v8h hi = *(const v8h*)(arow + 32 * kb + 16);
      v16h a = __builtin_shufflevector(lo, hi, 0, 1, 2, 3, 4, 5, 6, 7,
                                       8, 9, 10, 11, 12, 13, 14, 15);
      acc = __builtin_amdgcn_wmma_f32_16x16x32_f16(
          /*neg_a=*/false, a, /*neg_b=*/false, bfrag[kb],
          /*c_mod=*/(short)0, acc, /*reuse_a=*/false, /*reuse_b=*/false);
    }
    const long rbase = tileBase + rt * 16 + 8 * hh;
    float* orow = out + rbase * OUT_F + colg;   // row stride OUT_F floats
    if (fullTile) {
      // hot path: no per-store guards, streaming (non-temporal) stores
#pragma unroll
      for (int m = 0; m < 8; ++m)
        __builtin_nontemporal_store(acc[m], orow + (long)m * OUT_F);
    } else {
#pragma unroll
      for (int m = 0; m < 8; ++m) {
        if (rbase + m < N_ROWS)
          __builtin_nontemporal_store(acc[m], orow + (long)m * OUT_F);
      }
    }
  }
}

// ---------------------------------------------------------------------------
extern "C" void kernel_launch(void* const* d_in, const int* in_sizes, int n_in,
                              void* d_out, int out_size, void* d_ws, size_t ws_size,
                              hipStream_t stream) {
  (void)in_sizes; (void)n_in; (void)out_size; (void)ws_size;
  const float* t_in = (const float*)d_in[0];
  const float* o1   = (const float*)d_in[1];
  const float* o2   = (const float*)d_in[2];
  const float* W    = (const float*)d_in[3];
  const float* s    = (const float*)d_in[4];
  float* out = (float*)d_out;
  float* uv  = (float*)d_ws;   // 512 floats of scratch

  gat_prep<<<1, 256, 0, stream>>>(W, s, uv);

  const int numTiles = (N_ROWS + 127) / 128;
  const size_t smem = (size_t)128 * KPAD * sizeof(_Float16) + 512 * sizeof(float);
  hipFuncSetAttribute((const void*)gat_main,
                      hipFuncAttributeMaxDynamicSharedMemorySize, (int)smem);
  gat_main<<<numTiles, 256, smem, stream>>>(t_in, o1, o2, W, uv, out);
}
